// Net_7017976562059
// MI455X (gfx1250) — matrix operand order repaired and checked by reading, not compiled, not run on hardware
//
#include <hip/hip_runtime.h>
#include <hip/hip_bf16.h>
#include <hip/hip_fp16.h>

typedef __attribute__((ext_vector_type(16))) _Float16 v16h;
typedef __attribute__((ext_vector_type(8)))  _Float16 v8h;
typedef __attribute__((ext_vector_type(4)))  _Float16 v4h;
typedef __attribute__((ext_vector_type(8)))  float    v8f;

#define N_NODES 100000
#define N_EDGES 1600000
#define FDIM    128
#define NCLS    64

// ---------------- elementwise conversion f32 -> f16 ----------------
__global__ void k_f32_to_f16(const float* __restrict__ in, _Float16* __restrict__ out, int n) {
    int i = blockIdx.x * blockDim.x + threadIdx.x;
    if (i < n) out[i] = (_Float16)in[i];
}

// ---------------- degree / norm ----------------
__global__ void k_deg_init(float* __restrict__ deg, int n) {
    int i = blockIdx.x * blockDim.x + threadIdx.x;
    if (i < n) deg[i] = 1.0f;                      // self-loop contributes 1
}
__global__ void k_deg_count(const int* __restrict__ dst, float* __restrict__ deg, int e) {
    int i = blockIdx.x * blockDim.x + threadIdx.x;
    if (i < e) atomicAdd(&deg[dst[i]], 1.0f);
}
__global__ void k_dinv(float* __restrict__ deg, int n) {
    int i = blockIdx.x * blockDim.x + threadIdx.x;
    if (i < n) {
        float d = deg[i];
        deg[i] = (d > 0.0f) ? rsqrtf(d) : 0.0f;    // in-place deg -> dinv
    }
}

// ---------------- WMMA GEMM: D[M,OUT] = A[M,128] * W[OUT,128]^T ----------------
// One wave computes a 16x16 tile; block has OUT/16 waves (one per N-tile);
// grid.x = M/16 (M = 100000 is a multiple of 16).
// Fragment layouts used here (wave32, contiguous k = 16*hi + s, same order in A and B):
//   A 16x32 f16 : lane(m=lane&15,hi=lane>>4): halves 0..15 = A[row, k0+hi*16 .. +15]
//   B 32x16 f16 : lane(n=lane&15,hi):        halves 0..15 = W[ocol, k0+hi*16 .. +15]
//   C/D 16x16 f32: lane(n,hi): c[v] = D[8*hi+v, n]
template<int OUT, bool F16OUT>
__global__ void k_gemm(const _Float16* __restrict__ A, const _Float16* __restrict__ W,
                       void* __restrict__ Dst, const float* __restrict__ bias) {
    const int lane  = threadIdx.x & 31;
    const int ntile = threadIdx.x >> 5;
    const int mtile = blockIdx.x;
    const int m     = lane & 15;
    const int hi    = lane >> 4;
    const int row   = mtile * 16 + m;
    const int ocol  = ntile * 16 + m;       // B-fragment column = lane&15

    v8f c = {};
#pragma unroll
    for (int k0 = 0; k0 < FDIM; k0 += 32) {
        union { v16h v; v8h h[2]; } a;
        a.v = *(const v16h*)(A + (size_t)row * FDIM + k0 + hi * 16);
        v16h b = *(const v16h*)(W + (size_t)ocol * FDIM + k0 + hi * 16);
        c = __builtin_amdgcn_wmma_f32_16x16x32_f16(false, a.v, false, b,
                                                   (short)0, c, false, false);
    }

    const int n = lane & 15;
#pragma unroll
    for (int v = 0; v < 8; ++v) {
        int r  = mtile * 16 + 8 * hi + v;
        int oc = ntile * 16 + n;
        if (F16OUT) {
            ((_Float16*)Dst)[(size_t)r * OUT + oc] = (_Float16)c[v];
        } else {
            ((float*)Dst)[(size_t)r * OUT + oc] = c[v] + bias[oc];
        }
    }
}

// ---------------- accumulator init with self-loop message ----------------
// acc[i,c] = dinv[i]^2 * h[i,c]
__global__ void k_acc_self(const _Float16* __restrict__ H, const float* __restrict__ dinv,
                           float* __restrict__ acc, int nf) {
    int idx = blockIdx.x * blockDim.x + threadIdx.x;
    if (idx < nf) {
        float di = dinv[idx >> 7];              // node = idx / 128
        acc[idx] = di * di * (float)H[idx];
    }
}

// ---------------- edge aggregation: one wave per edge ----------------
// acc[dst] += dinv[src]*dinv[dst] * h[src]   (128 feats, 4 per lane, f16 gather, f32 atomics)
__global__ void k_agg(const _Float16* __restrict__ H, const int* __restrict__ src,
                      const int* __restrict__ dst, const float* __restrict__ dinv,
                      float* __restrict__ acc, int ne) {
    int gwave = (int)((blockIdx.x * (size_t)blockDim.x + threadIdx.x) >> 5);
    if (gwave >= ne) return;
    int lane = threadIdx.x & 31;
    int s = src[gwave];
    int d = dst[gwave];
    float nrm = dinv[s] * dinv[d];
    v4h hv = *(const v4h*)(H + (size_t)s * FDIM + lane * 4);
    float* ap = acc + (size_t)d * FDIM + lane * 4;
    atomicAdd(ap + 0, nrm * (float)hv[0]);
    atomicAdd(ap + 1, nrm * (float)hv[1]);
    atomicAdd(ap + 2, nrm * (float)hv[2]);
    atomicAdd(ap + 3, nrm * (float)hv[3]);
}

// ---------------- bias + relu + convert back to f16 (next layer input) ----------------
__global__ void k_relu_bias_cvt(const float* __restrict__ acc, const float* __restrict__ bias,
                                _Float16* __restrict__ out, int nf) {
    int idx = blockIdx.x * blockDim.x + threadIdx.x;
    if (idx < nf) {
        float v = acc[idx] + bias[idx & (FDIM - 1)];
        out[idx] = (_Float16)fmaxf(v, 0.0f);
    }
}

extern "C" void kernel_launch(void* const* d_in, const int* in_sizes, int n_in,
                              void* d_out, int out_size, void* d_ws, size_t ws_size,
                              hipStream_t stream) {
    const float* x  = (const float*)d_in[0];
    const int*   ei = (const int*)  d_in[1];   // [2, E] flat: row0 = src, row1 = dst
    const float* W1 = (const float*)d_in[2];
    const float* b1 = (const float*)d_in[3];
    const float* W2 = (const float*)d_in[4];
    const float* b2 = (const float*)d_in[5];
    const float* Wl = (const float*)d_in[6];
    const float* bl = (const float*)d_in[7];
    float* out = (float*)d_out;

    const int* srcp = ei;
    const int* dstp = ei + N_EDGES;

    // workspace layout (all regions 256B-aligned by construction)
    char* ws = (char*)d_ws;
    float*    dinv = (float*)(ws);                                   //  0.4 MB (deg -> dinv in place)
    _Float16* A16  = (_Float16*)(ws + (size_t)1  * 1024 * 1024);     // 25.6 MB layer input (f16)
    _Float16* H16  = (_Float16*)(ws + (size_t)28 * 1024 * 1024);     // 25.6 MB pre-agg h (f16)
    float*    ACC  = (float*)   (ws + (size_t)55 * 1024 * 1024);     // 51.2 MB f32 accumulator
    _Float16* W1h  = (_Float16*)(ws + (size_t)108 * 1024 * 1024);    // weights (f16)
    _Float16* W2h  = W1h + FDIM * FDIM;
    _Float16* Wlh  = W2h + FDIM * FDIM;

    const int nf = N_NODES * FDIM;                 // 12,800,000 (multiple of 256)

    // --- precompute: f16 copies of x and weights; degree norms ---
    k_f32_to_f16<<<nf / 256, 256, 0, stream>>>(x, A16, nf);
    k_f32_to_f16<<<(FDIM * FDIM) / 256, 256, 0, stream>>>(W1, W1h, FDIM * FDIM);
    k_f32_to_f16<<<(FDIM * FDIM) / 256, 256, 0, stream>>>(W2, W2h, FDIM * FDIM);
    k_f32_to_f16<<<(NCLS * FDIM) / 256, 256, 0, stream>>>(Wl, Wlh, NCLS * FDIM);

    k_deg_init <<<(N_NODES + 255) / 256, 256, 0, stream>>>(dinv, N_NODES);
    k_deg_count<<<N_EDGES / 256,        256, 0, stream>>>(dstp, dinv, N_EDGES);
    k_dinv     <<<(N_NODES + 255) / 256, 256, 0, stream>>>(dinv, N_NODES);

    const dim3 gtiles(N_NODES / 16);               // 6250 M-tiles

    // --- layer 1 ---
    k_gemm<FDIM, true><<<gtiles, 256, 0, stream>>>(A16, W1h, (void*)H16, nullptr);
    k_acc_self<<<nf / 256, 256, 0, stream>>>(H16, dinv, ACC, nf);
    k_agg<<<N_EDGES / 8, 256, 0, stream>>>(H16, srcp, dstp, dinv, ACC, N_EDGES);
    k_relu_bias_cvt<<<nf / 256, 256, 0, stream>>>(ACC, b1, A16, nf);

    // --- layer 2 ---
    k_gemm<FDIM, true><<<gtiles, 256, 0, stream>>>(A16, W2h, (void*)H16, nullptr);
    k_acc_self<<<nf / 256, 256, 0, stream>>>(H16, dinv, ACC, nf);
    k_agg<<<N_EDGES / 8, 256, 0, stream>>>(H16, srcp, dstp, dinv, ACC, N_EDGES);
    k_relu_bias_cvt<<<nf / 256, 256, 0, stream>>>(ACC, b2, A16, nf);

    // --- classifier head: out[N, 64] f32 = h2 @ Wl^T + bl ---
    k_gemm<NCLS, false><<<gtiles, 128, 0, stream>>>(A16, Wlh, (void*)out, bl);
}
